// QMDPLayer_81630148428369
// MI455X (gfx1250) — matrix operand order repaired and checked by reading, not compiled, not run on hardware
//
#include <hip/hip_runtime.h>

#define T_DIM 64
#define B_DIM 1024
#define S_DIM 256
#define A_DIM 16
#define H_DIM 32
#define R_DIM 32
#define KTOT   (A_DIM * S_DIM)   // 4096 = contraction length of belief GEMM
#define KSPLIT 4                 // K-split factor for GEMM parallelism
#define KCHUNK (KTOT / KSPLIT)   // 1024 per block

typedef __attribute__((ext_vector_type(16))) __bf16 v16bf;
typedef __attribute__((ext_vector_type(8)))  __bf16 v8bf;
typedef __attribute__((ext_vector_type(8)))  float  v8f;

union FragB16 { v16bf v; v8bf h[2]; };

__device__ __forceinline__ float warp_sum(float x) {
#pragma unroll
  for (int off = 16; off > 0; off >>= 1) x += __shfl_xor(x, off, 32);
  return x;
}
__device__ __forceinline__ float warp_max(float x) {
#pragma unroll
  for (int off = 16; off > 0; off >>= 1) x = fmaxf(x, __shfl_xor(x, off, 32));
  return x;
}

// ---------------------------------------------------------------------------
// Precompute 1: transT[j][k*S+i] = softmax_j( sum_r pu[r,i]*pv[r,j]*pw[r,k] )
// Stored bf16, K-contiguous (transposed) so GEMM B-fragments are contiguous.
// ---------------------------------------------------------------------------
__global__ __launch_bounds__(256) void build_transT(
    const float* __restrict__ pu, const float* __restrict__ pv,
    const float* __restrict__ pw, unsigned short* __restrict__ transT_us) {
  __bf16* transT = reinterpret_cast<__bf16*>(transT_us);
  const int lane = threadIdx.x & 31;
  const int wid  = threadIdx.x >> 5;
  const int row  = blockIdx.x * 8 + wid;   // row = k*S + i  (0..4095)
  const int k = row >> 8;
  const int i = row & 255;

  float pr = pu[lane * S_DIM + i] * pw[lane * A_DIM + k];

  float acc[8];
#pragma unroll
  for (int jj = 0; jj < 8; ++jj) acc[jj] = 0.f;
  for (int r = 0; r < R_DIM; ++r) {
    float s = __shfl(pr, r, 32);
#pragma unroll
    for (int jj = 0; jj < 8; ++jj)
      acc[jj] += s * pv[r * S_DIM + lane + jj * 32];
  }
  float mx = -3.4e38f;
#pragma unroll
  for (int jj = 0; jj < 8; ++jj) mx = fmaxf(mx, acc[jj]);
  mx = warp_max(mx);
  float sum = 0.f;
#pragma unroll
  for (int jj = 0; jj < 8; ++jj) { acc[jj] = __expf(acc[jj] - mx); sum += acc[jj]; }
  sum = warp_sum(sum);
  const float inv = 1.f / sum;
#pragma unroll
  for (int jj = 0; jj < 8; ++jj) {
    const int j = lane + jj * 32;
    transT[(size_t)j * KTOT + row] = (__bf16)(acc[jj] * inv);
  }
}

// ---------------------------------------------------------------------------
// Precompute 2: truncated Poisson pdf over horizons 1..32 (one wave).
// ---------------------------------------------------------------------------
__global__ void build_taupdf(const float* __restrict__ tau_param,
                             float* __restrict__ taupdf) {
  const int lane = threadIdx.x;
  const float tau = __expf(fminf(fmaxf(tau_param[0], -13.815510558f), 6.9077552790f));
  const int k = lane + 1;
  float lg = 0.f;                       // lgamma(k+1) = log(k!)
  for (int m = 2; m <= k; ++m) lg += __logf((float)m);
  float lp = (float)k * __logf(tau) - tau - lg;
  float mx = warp_max(lp);
  float e  = __expf(lp - mx);
  float s  = warp_sum(e);
  taupdf[lane] = e / s;
}

// ---------------------------------------------------------------------------
// Belief GEMM with K-split + double-buffered LDS staging.
//   s_next = bel_big[1024,4096](bf16) @ transT'[4096,256], fp32 acc.
//   bel_big[n, k*S+i] = u[n,k]*bel[n,i]  (built on the fly into LDS)
// Grid (8, 4, KSPLIT); block 256 = 8 waves; tile 128M x 64N; K-chunks of 32.
// B tile staged via global_load_async_to_lds_b128, A tile via cvt+ds_store.
// ---------------------------------------------------------------------------
__device__ __forceinline__ void stage_tiles(
    __bf16 (*As)[32], __bf16 (*Bst)[32],
    const float* __restrict__ bel, const float* __restrict__ u_t,
    const __bf16* __restrict__ transT, int m0, int n0, int kk, int tid)
{
  const int ar  = tid >> 1;          // A: row 0..127
  const int ac0 = (tid & 1) * 16;    // A: 16 contiguous cols
  const int br  = tid >> 2;          // B: row (j) 0..63
  const int bc0 = (tid & 3) * 8;     // B: 8 bf16 = 16 bytes

  // async stage B tile: Bst[j][kq] = transT[(n0+j)*4096 + kk + kq]
  {
    const __bf16* gsrc = transT + (size_t)(n0 + br) * KTOT + kk + bc0;
    unsigned lds_off = (unsigned)(unsigned long long)(&Bst[br][bc0]);
    asm volatile("global_load_async_to_lds_b128 %0, %1, off"
                 :: "v"(lds_off), "v"(gsrc) : "memory");
  }
  // stage A tile with on-the-fly scale + bf16 convert
  {
    const int   kact  = kk >> 8;       // action index of this K chunk
    const int   ibase = kk & 255;      // state offset of this K chunk
    const int   n     = m0 + ar;
    const float scale = u_t[n * A_DIM + kact];
    const float* bsrc = bel + (size_t)n * S_DIM + ibase + ac0;
#pragma unroll
    for (int c = 0; c < 16; ++c)
      As[ar][ac0 + c] = (__bf16)(scale * bsrc[c]);
  }
}

__global__ __launch_bounds__(256) void gemm_snext(
    const float* __restrict__ bel,                  // [B,S] fp32
    const float* __restrict__ u_t,                  // [B,A] fp32
    const unsigned short* __restrict__ transT_us,   // [256][4096] bf16
    float* __restrict__ spart)                      // [KSPLIT][B,S] fp32
{
  const __bf16* transT = reinterpret_cast<const __bf16*>(transT_us);
  __shared__ __bf16 As[2][128][32];   // M x Kc, double buffered
  __shared__ __bf16 Bst[2][64][32];   // N x Kc (K-contiguous), double buffered

  const int tid   = threadIdx.x;
  const int lane  = tid & 31;
  const int wid   = tid >> 5;
  const int waveM = wid >> 1;      // 0..3 -> 32-row slab
  const int waveN = wid & 1;       // 0..1 -> 32-col slab
  const int m0 = blockIdx.x * 128;
  const int n0 = blockIdx.y * 64;
  const int kbeg = blockIdx.z * KCHUNK;
  const int kend = kbeg + KCHUNK;

  v8f acc[2][2];
  const v8f vzero = {0.f,0.f,0.f,0.f,0.f,0.f,0.f,0.f};
#pragma unroll
  for (int tm = 0; tm < 2; ++tm)
#pragma unroll
    for (int tn = 0; tn < 2; ++tn) acc[tm][tn] = vzero;

  // prologue: fill buffer 0
  stage_tiles(As[0], Bst[0], bel, u_t, transT, m0, n0, kbeg, tid);
  asm volatile("s_wait_asynccnt 0x0" ::: "memory");
  __syncthreads();

  const int mrow  = lane & 15;
  const int koff  = (lane >= 16) ? 8 : 0;    // A: K halves split by half-wave
  const int kbase = (lane >= 16) ? 8 : 0;    // B: K halves split by half-wave
  const int ncol  = lane & 15;

  int cur = 0;
  for (int kk = kbeg; kk < kend; kk += 32) {
    // prefetch next chunk into the other buffer while computing this one
    if (kk + 32 < kend)
      stage_tiles(As[cur ^ 1], Bst[cur ^ 1], bel, u_t, transT, m0, n0, kk + 32, tid);

    FragB16 fa[2], fb[2];
#pragma unroll
    for (int tm = 0; tm < 2; ++tm) {
      const int mr = waveM * 32 + tm * 16 + mrow;
      fa[tm].h[0] = *reinterpret_cast<const v8bf*>(&As[cur][mr][koff]);       // K 0..7 / 8..15
      fa[tm].h[1] = *reinterpret_cast<const v8bf*>(&As[cur][mr][16 + koff]);  // K 16..23 / 24..31
    }
#pragma unroll
    for (int tn = 0; tn < 2; ++tn) {
      const int nr = waveN * 32 + tn * 16 + ncol;
      fb[tn].h[0] = *reinterpret_cast<const v8bf*>(&Bst[cur][nr][kbase]);
      fb[tn].h[1] = *reinterpret_cast<const v8bf*>(&Bst[cur][nr][kbase + 16]);
    }
#pragma unroll
    for (int tm = 0; tm < 2; ++tm)
#pragma unroll
      for (int tn = 0; tn < 2; ++tn)
        acc[tm][tn] = __builtin_amdgcn_wmma_f32_16x16x32_bf16(
            false, fa[tm].v, false, fb[tn].v, (short)0, acc[tm][tn],
            false, false);

    asm volatile("s_wait_asynccnt 0x0" ::: "memory");
    __syncthreads();
    cur ^= 1;
  }

  // store partial C/D (v8f: M = v + 8*(lane>=16), N = lane&15)
  float* out = spart + (size_t)blockIdx.z * B_DIM * S_DIM;
  const int rsel = (lane >= 16) ? 8 : 0;
#pragma unroll
  for (int tm = 0; tm < 2; ++tm)
#pragma unroll
    for (int tn = 0; tn < 2; ++tn)
#pragma unroll
      for (int v = 0; v < 8; ++v) {
        const int row = m0 + waveM * 32 + tm * 16 + v + rsel;
        const int col = n0 + waveN * 32 + tn * 16 + (lane & 15);
        out[(size_t)row * S_DIM + col] = acc[tm][tn][v];
      }
}

// ---------------------------------------------------------------------------
// b_post = softmax(log(sum_z spart[z] + eps) + logp_o[t]) : one wave per row.
// ---------------------------------------------------------------------------
__global__ __launch_bounds__(256) void bpost_softmax(
    const float* __restrict__ spart, const float* __restrict__ lo_t,
    float* __restrict__ bpost_out) {
  const int lane = threadIdx.x & 31;
  const int n = blockIdx.x * 8 + (threadIdx.x >> 5);
  float x[8];
  float mx = -3.4e38f;
#pragma unroll
  for (int jj = 0; jj < 8; ++jj) {
    const size_t idx = (size_t)n * S_DIM + lane + jj * 32;
    float p = 0.f;
#pragma unroll
    for (int z = 0; z < KSPLIT; ++z)
      p += spart[(size_t)z * B_DIM * S_DIM + idx];
    x[jj] = __logf(p + 1e-6f) + lo_t[idx];
    mx = fmaxf(mx, x[jj]);
  }
  mx = warp_max(mx);
  float s = 0.f;
#pragma unroll
  for (int jj = 0; jj < 8; ++jj) { x[jj] = __expf(x[jj] - mx); s += x[jj]; }
  s = warp_sum(s);
  const float inv = 1.f / s;
#pragma unroll
  for (int jj = 0; jj < 8; ++jj)
    bpost_out[(size_t)n * S_DIM + lane + jj * 32] = x[jj] * inv;
}

// ---------------------------------------------------------------------------
// q/pi: streams value[H,B,A,S] (HBM-bound, the roofline-dominant kernel).
// Block = 1024 threads = 32 waves, one block per batch n; wave h computes
// q[h,n,:], softmax_k, weight by tau_pdf[h], cross-wave LDS reduce -> pi[n,:].
// ---------------------------------------------------------------------------
__global__ __launch_bounds__(1024) void qpi_kernel(
    const float* __restrict__ bpost,   // [B,S]
    const float* __restrict__ value,   // [H,B,A,S]
    const float* __restrict__ taupdf,  // [H]
    float* __restrict__ pi_out)        // [B,A]
{
  __shared__ float bp[S_DIM];
  __shared__ float red[H_DIM][A_DIM];
  const int n    = blockIdx.x;
  const int tid  = threadIdx.x;
  const int lane = tid & 31;
  const int h    = tid >> 5;

  if (tid < S_DIM) bp[tid] = bpost[(size_t)n * S_DIM + tid];
  __syncthreads();

  float bv[8];
#pragma unroll
  for (int c = 0; c < 8; ++c) bv[c] = bp[lane * 8 + c];

  const float* vbase = value + (((size_t)h * B_DIM + n) * A_DIM) * S_DIM;
  float q[A_DIM];
#pragma unroll
  for (int k = 0; k < A_DIM; ++k) {
    const float* vr = vbase + (size_t)k * S_DIM + lane * 8;
    if (k + 1 < A_DIM) __builtin_prefetch(vr + S_DIM, 0, 0);  // global_prefetch_b8
    float a = 0.f;
#pragma unroll
    for (int c = 0; c < 8; ++c) a += bv[c] * vr[c];
    q[k] = warp_sum(a);                // all lanes hold q[h,n,k]
  }
  float mx = q[0];
#pragma unroll
  for (int k = 1; k < A_DIM; ++k) mx = fmaxf(mx, q[k]);
  float s = 0.f;
#pragma unroll
  for (int k = 0; k < A_DIM; ++k) { q[k] = __expf(q[k] - mx); s += q[k]; }
  const float w = taupdf[h] / s;
  if (lane < A_DIM) red[h][lane] = q[lane] * w;
  __syncthreads();
  if (tid < A_DIM) {
    float acc = 0.f;
#pragma unroll
    for (int hh = 0; hh < H_DIM; ++hh) acc += red[hh][tid];
    pi_out[(size_t)n * A_DIM + tid] = acc;
  }
}

// ---------------------------------------------------------------------------
extern "C" void kernel_launch(void* const* d_in, const int* in_sizes, int n_in,
                              void* d_out, int out_size, void* d_ws, size_t ws_size,
                              hipStream_t stream) {
  (void)in_sizes; (void)n_in; (void)out_size; (void)ws_size;
  const float* logp_o = (const float*)d_in[0];   // [T,B,S]
  const float* u      = (const float*)d_in[1];   // [T,B,A]
  const float* value  = (const float*)d_in[2];   // [H,B,A,S]
  const float* b      = (const float*)d_in[3];   // [B,S]
  const float* tau_p  = (const float*)d_in[4];   // [1,1]
  const float* pu     = (const float*)d_in[5];   // [1,R,S]
  const float* pv     = (const float*)d_in[6];   // [1,R,S]
  const float* pw     = (const float*)d_in[7];   // [1,R,A]

  float* alpha_b  = (float*)d_out;                                    // [T,B,S]
  float* alpha_pi = alpha_b + (size_t)T_DIM * B_DIM * S_DIM;          // [T,B,A]

  char* ws = (char*)d_ws;
  unsigned short* transT = (unsigned short*)ws;              // 2 MB bf16 [256][4096]
  float* spart  = (float*)(ws + (2u << 20));                 // 4 MB fp32 [KSPLIT][1024][256]
  float* taupdf = (float*)(ws + (6u << 20));                 // 128 B

  build_transT<<<dim3(KTOT / 8), dim3(256), 0, stream>>>(pu, pv, pw, transT);
  build_taupdf<<<dim3(1), dim3(32), 0, stream>>>(tau_p, taupdf);

  for (int t = 0; t < T_DIM; ++t) {
    const float* bel = (t == 0) ? b : (alpha_b + (size_t)(t - 1) * B_DIM * S_DIM);
    gemm_snext<<<dim3(B_DIM / 128, S_DIM / 64, KSPLIT), dim3(256), 0, stream>>>(
        bel, u + (size_t)t * B_DIM * A_DIM, transT, spart);
    bpost_softmax<<<dim3(B_DIM / 8), dim3(256), 0, stream>>>(
        spart, logp_o + (size_t)t * B_DIM * S_DIM,
        alpha_b + (size_t)t * B_DIM * S_DIM);
    qpi_kernel<<<dim3(B_DIM), dim3(1024), 0, stream>>>(
        alpha_b + (size_t)t * B_DIM * S_DIM, value, taupdf,
        alpha_pi + (size_t)t * B_DIM * A_DIM);
  }
}
